// GAT_82669530514080
// MI455X (gfx1250) — matrix operand order repaired and checked by reading, not compiled, not run on hardware
//
#include <hip/hip_runtime.h>

// ---------------- problem constants (match reference) ----------------
constexpr int    GN     = 8192;
constexpr int    F_IN   = 512;
constexpr int    F_HID  = 256;
constexpr float  ALPHA  = 0.2f;
constexpr float  NEGINF = -9e15f;

typedef __attribute__((ext_vector_type(16))) __bf16 v16bf;
typedef __attribute__((ext_vector_type(8)))  __bf16 v8bf;
typedef __attribute__((ext_vector_type(8)))  float  v8f;

// f32 -> bf16 round-to-nearest-even
static __device__ __forceinline__ __bf16 f2bf(float f) {
    unsigned u = __builtin_bit_cast(unsigned, f);
    u += 0x7FFFu + ((u >> 16) & 1u);
    unsigned short h = (unsigned short)(u >> 16);
    return __builtin_bit_cast(__bf16, h);
}

// Packed "B-operand" layout: element (k, col) of a [K x F] matrix lives at
//   ((k/32)*F + col)*32 + ((k/16)%2)*16 + (k%16)
// so the 16 bf16 values one lane feeds to v_wmma_*_bf16 as B are contiguous
// (32 bytes, 32B-aligned).
static __device__ __forceinline__ size_t pk_idx(int k, int col, int F) {
    return ((size_t)(k >> 5) * F + col) * 32 + ((k >> 4) & 1) * 16 + (k & 15);
}

// ---------------- elementwise f32 -> bf16 (row-major, A operand) --------
__global__ __launch_bounds__(256) void cvt_bf16_kernel(const float* __restrict__ src,
                                                       __bf16* __restrict__ dst, int n) {
    int i = blockIdx.x * 256 + threadIdx.x;
    if (i < n) dst[i] = f2bf(src[i]);
}

// ---------------- f32 -> packed-bf16 B layout (for weight matrices) -----
__global__ __launch_bounds__(256) void pack_b_kernel(const float* __restrict__ W,
                                                     __bf16* __restrict__ Bp,
                                                     int K, int F) {
    int i = blockIdx.x * 256 + threadIdx.x;
    if (i < K * F) {
        int k = i / F, f = i - k * F;
        Bp[pk_idx(k, f, F)] = f2bf(W[i]);
    }
}

// ---------------- WMMA GEMM: C[M,F] = A[M,K] @ B[K,F] -------------------
// A: row-major bf16.  Bp: packed-B bf16.  C: f32 row-major.
// Cb: packed-B bf16 (feeds the attention kernel as its B operand).
// One wave per 16x16 output tile; 8 waves/block -> 16 x 128 per block.
__global__ __launch_bounds__(256) void gemm_bf16_kernel(const __bf16* __restrict__ A,
                                                        const __bf16* __restrict__ Bp,
                                                        float* __restrict__ C,
                                                        __bf16* __restrict__ Cb,
                                                        int M, int K, int F) {
    const int lane = threadIdx.x & 31;
    const int wave = threadIdx.x >> 5;
    const int hi   = lane >> 4;           // 0: lanes 0-15, 1: lanes 16-31
    const int row0 = blockIdx.x * 16;
    const int col0 = (blockIdx.y * 8 + wave) * 16;
    if (col0 >= F) return;

    v8f acc = {};
    const int arow = row0 + (lane & 15);
    const int bcol = col0 + (lane & 15);

    for (int k0 = 0; k0 < K; k0 += 32) {
        // A fragment: rows of 8 contiguous bf16 -> two aligned b128 loads
        const __bf16* ap = A + (size_t)arow * K + k0 + hi * 16;
        v8bf alo = *(const v8bf*)(ap);
        v8bf ahi = *(const v8bf*)(ap + 8);
        v16bf a = __builtin_shufflevector(alo, ahi, 0, 1, 2, 3, 4, 5, 6, 7,
                                          8, 9, 10, 11, 12, 13, 14, 15);
        // B fragment: 16 contiguous bf16 in packed layout (32B aligned)
        const __bf16* bp = Bp + ((size_t)(k0 >> 5) * F + bcol) * 32 + hi * 16;
        v16bf b = *(const v16bf*)bp;
        acc = __builtin_amdgcn_wmma_f32_16x16x32_bf16(false, a, false, b,
                                                      (short)0, acc, false, false);
    }

    // f32 store (row-major): rows r + 8*hi, col = bcol
    #pragma unroll
    for (int r = 0; r < 8; ++r)
        C[(size_t)(row0 + r + 8 * hi) * F + bcol] = acc[r];

    // packed-bf16 store: the 8 rows land contiguously -> single b128 store
    v8bf cb;
    #pragma unroll
    for (int r = 0; r < 8; ++r) cb[r] = f2bf(acc[r]);
    size_t pkb = ((size_t)(row0 >> 5) * F + bcol) * 32 + ((row0 >> 4) & 1) * 16 + hi * 8;
    *(v8bf*)(Cb + pkb) = cb;
}

// ---------------- e1[i] = Wh[i,:].a_left, e2[i] = Wh[i,:].a_right -------
__global__ __launch_bounds__(256) void compute_e_kernel(const float* __restrict__ Wh,
                                                        const float* __restrict__ a,
                                                        float* __restrict__ e1,
                                                        float* __restrict__ e2, int Fout) {
    const int lane = threadIdx.x & 31;
    const int row  = blockIdx.x * 8 + (threadIdx.x >> 5);
    float s1 = 0.f, s2 = 0.f;
    for (int f = lane; f < Fout; f += 32) {
        float w = Wh[(size_t)row * Fout + f];
        s1 += w * a[f];
        s2 += w * a[Fout + f];
    }
    #pragma unroll
    for (int m = 16; m > 0; m >>= 1) {
        s1 += __shfl_xor(s1, m, 32);
        s2 += __shfl_xor(s2, m, 32);
    }
    if (lane == 0) { e1[row] = s1; e2[row] = s2; }
}

// ---------------- fused masked-softmax attention + att@Wh + ELU ---------
// Block: 16 graph rows, 8 waves splitting FOUT columns.
// Online (flash) softmax over j in chunks of 32 (= WMMA K for bf16).
// adj/e2 are software-pipelined one chunk ahead in registers; probabilities
// are staged in LDS in A-fragment order; Whp is in packed-B layout.
template <int FOUT>
__global__ __launch_bounds__(256) void gat_attn_kernel(const float* __restrict__ e1,
                                                       const float* __restrict__ e2,
                                                       const float* __restrict__ adj,
                                                       const __bf16* __restrict__ Whp,
                                                       float* __restrict__ out,
                                                       __bf16* __restrict__ outb) {
    constexpr int CPW   = FOUT / 8;   // columns per wave
    constexpr int TILES = CPW / 16;   // 16-wide N tiles per wave

    __shared__ float s_p[16][33];                 // raw masked scores
    __shared__ __align__(32) __bf16 s_a[32 * 16]; // probs in A-fragment order
    __shared__ __align__(16) float s_scale[16];
    __shared__ __align__(16) float s_l[16];
    __shared__ float s_m[16], s_e1[16];

    const int tid  = threadIdx.x;
    const int lane = tid & 31;
    const int wave = tid >> 5;
    const int hi   = lane >> 4;
    const int row0 = blockIdx.x * 16;
    const int colbase = wave * CPW;

    if (tid < 16) {
        s_m[tid]  = NEGINF;  // finite sentinel == reference mask value
        s_l[tid]  = 0.f;
        s_e1[tid] = e1[row0 + tid];
    }
    __syncthreads();

    // per-thread fixed score coordinates: entries (tid) and (tid+256) of 16x32
    const int r0 = tid >> 5,           jj0 = tid & 31;
    const int r1 = (tid + 256) >> 5,   jj1 = tid & 31;
    const float* adjp0 = adj + (size_t)(row0 + r0) * GN + jj0;
    const float* adjp1 = adj + (size_t)(row0 + r1) * GN + jj1;

    // pipeline prologue: chunk 0 operands in registers
    float av0 = adjp0[0], av1 = adjp1[0];
    float ev0 = e2[jj0],  ev1 = e2[jj1];

    v8f acc[TILES];
    #pragma unroll
    for (int t = 0; t < TILES; ++t) acc[t] = (v8f){};

    for (int jc = 0; jc < GN; jc += 32) {
        // ---- phase 1: masked leaky-relu scores from pipelined registers ----
        {
            float raw0 = s_e1[r0] + ev0;
            float lr0  = raw0 > 0.f ? raw0 : ALPHA * raw0;
            s_p[r0][jj0] = (av0 > 0.f) ? lr0 : NEGINF;
            float raw1 = s_e1[r1] + ev1;
            float lr1  = raw1 > 0.f ? raw1 : ALPHA * raw1;
            s_p[r1][jj1] = (av1 > 0.f) ? lr1 : NEGINF;
        }
        // issue next chunk's loads now; consumed after three barriers
        if (jc + 32 < GN) {
            av0 = adjp0[jc + 32];
            av1 = adjp1[jc + 32];
            ev0 = e2[jc + 32 + jj0];
            ev1 = e2[jc + 32 + jj1];
        }
        __syncthreads();

        // ---- phase 2: online softmax on wave 0 (2 lanes per row) ----------
        if (wave == 0) {
            const int r    = lane & 15;
            const int half = hi;               // jj base = 16*half
            float v[16];
            #pragma unroll
            for (int t = 0; t < 16; ++t) v[t] = s_p[r][half * 16 + t];
            float cm = NEGINF;
            #pragma unroll
            for (int t = 0; t < 16; ++t) cm = fmaxf(cm, v[t]);
            cm = fmaxf(cm, __shfl_xor(cm, 16, 32));
            float m_old = s_m[r];
            float m_new = fmaxf(m_old, cm);
            float sum = 0.f;
            #pragma unroll
            for (int t = 0; t < 16; ++t) {
                float p = __expf(v[t] - m_new);
                sum += p;
                int jj = half * 16 + t;
                int ln = r + (((jj >> 4) & 1) << 4);        // consuming lane
                int e  = (jj & 15);                         // element in frag
                s_a[ln * 16 + e] = f2bf(p);
            }
            sum += __shfl_xor(sum, 16, 32);
            if (half == 0) {
                float scale = __expf(m_old - m_new);
                s_scale[r] = scale;
                s_m[r]     = m_new;
                s_l[r]     = s_l[r] * scale + sum;
            }
        } else if (tid >= 224 && tid < 240 && jc + 96 < GN) {
            // wave-7 threads prefetch the adj stream three chunks ahead
            __builtin_prefetch(&adj[(size_t)(row0 + (tid - 224)) * GN + jc + 96], 0, 0);
        }
        __syncthreads();

        // ---- phase 3: rescale accumulators, WMMA P @ Wh --------------------
        float4 sc0 = *(const float4*)(&s_scale[8 * hi]);
        float4 sc1 = *(const float4*)(&s_scale[8 * hi + 4]);
        float sc[8] = {sc0.x, sc0.y, sc0.z, sc0.w, sc1.x, sc1.y, sc1.z, sc1.w};
        #pragma unroll
        for (int t = 0; t < TILES; ++t)
            #pragma unroll
            for (int r = 0; r < 8; ++r) acc[t][r] *= sc[r];

        // A fragment: one aligned 32-byte LDS load
        v16bf afrag = *(const v16bf*)(&s_a[lane * 16]);

        // B fragments: 32B contiguous loads from packed Wh (L2-resident)
        #pragma unroll
        for (int t = 0; t < TILES; ++t) {
            const int bcol = colbase + t * 16 + (lane & 15);
            const __bf16* bp = Whp + ((size_t)(jc >> 5) * FOUT + bcol) * 32 + hi * 16;
            v16bf bfrag = *(const v16bf*)bp;
            acc[t] = __builtin_amdgcn_wmma_f32_16x16x32_bf16(false, afrag, false, bfrag,
                                                             (short)0, acc[t], false, false);
        }
        __syncthreads();   // protect s_p / s_a before next chunk overwrites
    }

    // ---- epilogue: normalize, ELU, store ----
    float4 l0 = *(const float4*)(&s_l[8 * hi]);
    float4 l1 = *(const float4*)(&s_l[8 * hi + 4]);
    float lv[8] = {l0.x, l0.y, l0.z, l0.w, l1.x, l1.y, l1.z, l1.w};
    float linv[8];
    #pragma unroll
    for (int r = 0; r < 8; ++r) linv[r] = 1.0f / lv[r];

    #pragma unroll
    for (int t = 0; t < TILES; ++t) {
        const int bcol = colbase + t * 16 + (lane & 15);
        #pragma unroll
        for (int r = 0; r < 8; ++r) {
            float v = acc[t][r] * linv[r];
            float o = v > 0.f ? v : (__expf(v) - 1.0f);   // ELU(alpha=1)
            size_t off = (size_t)(row0 + r + 8 * hi) * FOUT + bcol;
            out[off] = o;
            if (outb) outb[off] = f2bf(o);
        }
    }
}

// ---------------- host orchestration ----------------
extern "C" void kernel_launch(void* const* d_in, const int* in_sizes, int n_in,
                              void* d_out, int out_size, void* d_ws, size_t ws_size,
                              hipStream_t stream) {
    (void)in_sizes; (void)n_in; (void)out_size; (void)ws_size;

    const float* x   = (const float*)d_in[0];   // [N, F_IN]
    const float* adj = (const float*)d_in[1];   // [N, N]
    const float* W1  = (const float*)d_in[2];   // [F_IN, F_HID]
    const float* a1  = (const float*)d_in[3];   // [2*F_HID]
    const float* W2  = (const float*)d_in[4];   // [F_HID, F_IN]
    const float* a2  = (const float*)d_in[5];   // [2*F_IN]

    float* dec = (float*)d_out;                        // [N, F_IN]
    float* enc = dec + (size_t)GN * F_IN;              // [N, F_HID]

    // workspace layout
    char* ws = (char*)d_ws;
    size_t off = 0;
    auto alloc = [&](size_t bytes) { char* p = ws + off; off += (bytes + 255) & ~(size_t)255; return p; };
    __bf16* xb   = (__bf16*)alloc((size_t)GN * F_IN * 2);    // row-major (A op)
    __bf16* W1b  = (__bf16*)alloc((size_t)F_IN * F_HID * 2); // packed B
    __bf16* W2b  = (__bf16*)alloc((size_t)F_HID * F_IN * 2); // packed B
    float*  Wh1  = (float*) alloc((size_t)GN * F_HID * 4);
    __bf16* Wh1b = (__bf16*)alloc((size_t)GN * F_HID * 2);   // packed B
    __bf16* encb = (__bf16*)alloc((size_t)GN * F_HID * 2);   // row-major (A op)
    float*  Wh2  = (float*) alloc((size_t)GN * F_IN * 4);
    __bf16* Wh2b = (__bf16*)alloc((size_t)GN * F_IN * 2);    // packed B
    float*  e1a  = (float*) alloc((size_t)GN * 4);
    float*  e2a  = (float*) alloc((size_t)GN * 4);
    float*  e1b  = (float*) alloc((size_t)GN * 4);
    float*  e2b  = (float*) alloc((size_t)GN * 4);

    // ---- conversions / packing ----
    {
        int n = GN * F_IN;
        cvt_bf16_kernel<<<n / 256, 256, 0, stream>>>(x, xb, n);
        n = F_IN * F_HID;
        pack_b_kernel<<<n / 256, 256, 0, stream>>>(W1, W1b, F_IN, F_HID);
        n = F_HID * F_IN;
        pack_b_kernel<<<n / 256, 256, 0, stream>>>(W2, W2b, F_HID, F_IN);
    }

    // ---- layer 1: Wh1 = x @ W1 ----
    gemm_bf16_kernel<<<dim3(GN / 16, F_HID / 128), 256, 0, stream>>>(
        xb, W1b, Wh1, Wh1b, GN, F_IN, F_HID);
    compute_e_kernel<<<GN / 8, 256, 0, stream>>>(Wh1, a1, e1a, e2a, F_HID);
    gat_attn_kernel<F_HID><<<GN / 16, 256, 0, stream>>>(e1a, e2a, adj, Wh1b, enc, encb);

    // ---- layer 2: Wh2 = enc @ W2 ----
    gemm_bf16_kernel<<<dim3(GN / 16, F_IN / 128), 256, 0, stream>>>(
        encb, W2b, Wh2, Wh2b, GN, F_HID, F_IN);
    compute_e_kernel<<<GN / 8, 256, 0, stream>>>(Wh2, a2, e1b, e2b, F_IN);
    gat_attn_kernel<F_IN><<<GN / 16, 256, 0, stream>>>(e1b, e2b, adj, Wh2b, dec, nullptr);
}
